// KernelAttention_58720792871833
// MI455X (gfx1250) — matrix operand order repaired and checked by reading, not compiled, not run on hardware
//
#include <hip/hip_runtime.h>

// ---------------------------------------------------------------------------
// Linear attention (axial RoPE, elu+1 feature map) for MI455X / gfx1250.
// Heavy matmuls: v_wmma_f32_16x16x32_bf16 (wave32 WMMA).
// Tile staging: GLOBAL_LOAD_ASYNC_TO_LDS_B128 (async-to-LDS, ASYNCcnt),
// plus global_prefetch of the next k-tile.
// ---------------------------------------------------------------------------

typedef __attribute__((ext_vector_type(16))) __bf16 v16bf;
typedef __attribute__((ext_vector_type(8)))  float  v8f;

#define TOKENS   16384
#define BATCH    4
#define CH       512
#define HEADS    8
#define HDIM     64
#define BN_TOK   (BATCH * TOKENS)   // 65536

// ---- gfx1250 async global->LDS probe --------------------------------------
#if defined(__has_builtin)
#if __has_builtin(__builtin_amdgcn_global_load_async_to_lds_b128) && \
    __has_builtin(__builtin_amdgcn_s_wait_asynccnt)
#define USE_ASYNC_LDS 1
#endif
#endif
#ifndef USE_ASYNC_LDS
#define USE_ASYNC_LDS 0
#endif

#if USE_ASYNC_LDS
// Builtin signature (from diagnostics): (v4i addrspace(1)*, v4i addrspace(3)*, Ii, Ii)
typedef int v4i_async __attribute__((__vector_size__(4 * sizeof(int))));
typedef __attribute__((address_space(1))) v4i_async* gptr128_t;
typedef __attribute__((address_space(3))) v4i_async* lptr128_t;
#define AS1G(p) ((gptr128_t)(p))
#define AS3L(p) ((lptr128_t)(p))
#endif

// ======================= f32 -> bf16 conversion ============================
__global__ __launch_bounds__(256) void cvt_f32_bf16(const float* __restrict__ s,
                                                    __bf16* __restrict__ d, int n) {
    int i = blockIdx.x * 256 + threadIdx.x;
    if (i < n) d[i] = (__bf16)s[i];
}

// ======================= generic bf16 WMMA GEMM ============================
// C[M,N] = A[M,K] @ B[K,N] + bias[N].  M%64==0, N%128==0, K%32==0.
// 256 threads = 8 waves in 2(M) x 4(N); each wave owns 32x32 C (2x2 WMMA).
// A staged row-major (async-to-LDS b128, rows 16B aligned),
// B staged transposed so both fragment gathers are aligned b32 LDS loads.
template <bool OUTF32>
__global__ __launch_bounds__(256)
void gemm_bf16_kernel(const __bf16* __restrict__ A, const __bf16* __restrict__ Bm,
                      const float* __restrict__ bias, void* __restrict__ Cout,
                      int M, int N, int K) {
    constexpr int BM = 64, BN = 128, BK = 32, LKB = BK + 2;
    __shared__ __bf16 As[BM][BK];    // stride 64B: b128 rows stay 16B aligned
    __shared__ __bf16 Bs[BN][LKB];

    const int tid  = threadIdx.x;
    const int lane = tid & 31, wave = tid >> 5;
    const int wm = wave >> 2, wn = wave & 3;
    const int lrow = lane & 15, lhi = lane >> 4;
    const int n0 = blockIdx.x * BN;
    const int m0 = blockIdx.y * BM;

    v8f acc[2][2];
#pragma unroll
    for (int tm = 0; tm < 2; ++tm)
#pragma unroll
        for (int tn = 0; tn < 2; ++tn)
#pragma unroll
            for (int i = 0; i < 8; ++i) acc[tm][tn][i] = 0.f;

    for (int k0 = 0; k0 < K; k0 += BK) {
        // ---- stage A tile (64x32 bf16 = 4KB): one b128 per thread ----
        {
            const int idx = tid * 8;
            const int r = idx >> 5, c = idx & 31;
            const __bf16* src = A + (size_t)(m0 + r) * K + k0 + c;
#if USE_ASYNC_LDS
            __builtin_amdgcn_global_load_async_to_lds_b128(
                AS1G(src), AS3L(&As[r][c]), 0, 0);
#else
            const unsigned long long* s2 = (const unsigned long long*)src;
            unsigned long long d0 = s2[0], d1 = s2[1];
            *(unsigned int*)&As[r][c]     = (unsigned int)d0;
            *(unsigned int*)&As[r][c + 2] = (unsigned int)(d0 >> 32);
            *(unsigned int*)&As[r][c + 4] = (unsigned int)d1;
            *(unsigned int*)&As[r][c + 6] = (unsigned int)(d1 >> 32);
#endif
        }
        // ---- stage B tile (32x128) transposed into Bs[n][k], k-pairs ----
#pragma unroll
        for (int it = 0; it < 2; ++it) {
            const int idx = tid + it * 256;       // 0..511
            const int r = (idx >> 5) * 2;         // k row pair
            const int c = (idx & 31) * 4;         // 4 consecutive n
            union { unsigned long long u; __bf16 h[4]; } t0, t1;
            t0.u = *(const unsigned long long*)(Bm + (size_t)(k0 + r) * N + n0 + c);
            t1.u = *(const unsigned long long*)(Bm + (size_t)(k0 + r + 1) * N + n0 + c);
#pragma unroll
            for (int j = 0; j < 4; ++j) {
                union { unsigned int u; __bf16 h[2]; } w;
                w.h[0] = t0.h[j];
                w.h[1] = t1.h[j];
                *(unsigned int*)&Bs[c + j][r] = w.u;
            }
        }
        // ---- prefetch next k-tile into cache (global_prefetch_b8) ----
        if (k0 + BK < K) {
            const int ia = tid * 8;
            __builtin_prefetch(A + (size_t)(m0 + (ia >> 5)) * K + k0 + BK + (ia & 31), 0, 0);
            const int ib = tid * 16;
            __builtin_prefetch(Bm + (size_t)(k0 + BK + (ib >> 7)) * N + n0 + (ib & 127), 0, 0);
        }
#if USE_ASYNC_LDS
        __builtin_amdgcn_s_wait_asynccnt(0);
#endif
        __syncthreads();

        // ---- fragment gathers A(16x32)/B(32x16), both in contiguous k = 16*lhi + s order ----
        union Uv { unsigned int u[8]; v16bf v; };
        Uv afr[2], bfr[2];
#pragma unroll
        for (int tm = 0; tm < 2; ++tm) {
            const int ar = wm * 32 + tm * 16 + lrow;
            const int kb = lhi * 16;
#pragma unroll
            for (int v = 0; v < 8; ++v) {
                const int kk = v * 2 + kb;
                afr[tm].u[v] = *(const unsigned int*)&As[ar][kk];
            }
        }
#pragma unroll
        for (int tn = 0; tn < 2; ++tn) {
            const int bc = wn * 32 + tn * 16 + lrow;
            const int kb = lhi * 16;
#pragma unroll
            for (int v = 0; v < 8; ++v)
                bfr[tn].u[v] = *(const unsigned int*)&Bs[bc][kb + v * 2];
        }
#pragma unroll
        for (int tm = 0; tm < 2; ++tm)
#pragma unroll
            for (int tn = 0; tn < 2; ++tn)
                acc[tm][tn] = __builtin_amdgcn_wmma_f32_16x16x32_bf16(
                    false, afr[tm].v, false, bfr[tn].v,
                    (short)0, acc[tm][tn], false, false);
        __syncthreads();
    }

    // ---- epilogue: bias + store (C/D: lane 0-15 N=lane M=i; 16-31 M=8+i) ----
#pragma unroll
    for (int tm = 0; tm < 2; ++tm)
#pragma unroll
        for (int tn = 0; tn < 2; ++tn) {
            const int col = n0 + wn * 32 + tn * 16 + lrow;
            const float bv = bias ? bias[col] : 0.f;
#pragma unroll
            for (int i = 0; i < 8; ++i) {
                const int row = m0 + wm * 32 + tm * 16 + lhi * 8 + i;
                const float val = acc[tm][tn][i] + bv;
                if constexpr (OUTF32)
                    ((float*)Cout)[(size_t)row * N + col] = val;
                else
                    ((__bf16*)Cout)[(size_t)row * N + col] = (__bf16)val;
            }
        }
}

// ======================= RoPE + elu(.)+1, head-major split =================
// qkvh: [bn][3*512] bf16  ->  qh/kh/vh: [b*H+h][n][64] bf16
__global__ __launch_bounds__(256)
void rope_elu_kernel(const __bf16* __restrict__ qkvh, __bf16* __restrict__ qh,
                     __bf16* __restrict__ kh, __bf16* __restrict__ vh) {
    const int gid = blockIdx.x * 256 + threadIdx.x;  // BN_TOK*H*32 threads
    const int p  = gid & 31;          // pair index in head dim
    const int h  = (gid >> 5) & 7;
    const int bn = gid >> 8;
    const int b  = bn >> 14;
    const int t  = bn & (TOKENS - 1);
    const int tx = t & 127, ty = t >> 7;

    const int half = p >> 4;          // 0: x-angles, 1: y-angles
    const int j    = p & 15;          // frequency index
    const float freq = __expf(-(float)j * (2.3025851f / 16.f));  // 10^(-j/16)
    const float ang  = (half ? (float)ty : (float)tx) * freq;
    float cs, sn;
    __sincosf(ang, &sn, &cs);

    const size_t src = (size_t)bn * (3 * CH) + h * HDIM + half * 32 + 2 * j;
    const float qr = (float)qkvh[src],        qi = (float)qkvh[src + 1];
    const float kr = (float)qkvh[src + 512],  ki = (float)qkvh[src + 513];
    const float vr = (float)qkvh[src + 1024], vi = (float)qkvh[src + 1025];

    float qor = qr * cs - qi * sn, qoi = qr * sn + qi * cs;
    float kor = kr * cs - ki * sn, koi = kr * sn + ki * cs;
    // elu(x)+1
    qor = (qor > 0.f) ? qor + 1.f : __expf(qor);
    qoi = (qoi > 0.f) ? qoi + 1.f : __expf(qoi);
    kor = (kor > 0.f) ? kor + 1.f : __expf(kor);
    koi = (koi > 0.f) ? koi + 1.f : __expf(koi);

    const size_t dst = (((size_t)(b * HEADS + h)) * TOKENS + t) * HDIM + half * 32 + 2 * j;
    qh[dst] = (__bf16)qor;  qh[dst + 1] = (__bf16)qoi;
    kh[dst] = (__bf16)kor;  kh[dst + 1] = (__bf16)koi;
    vh[dst] = (__bf16)vr;   vh[dst + 1] = (__bf16)vi;
}

// ======================= kv = k^T v and ksum, split-N + atomics ============
__global__ __launch_bounds__(256)
void kv_reduce_kernel(const __bf16* __restrict__ kh, const __bf16* __restrict__ vh,
                      float* __restrict__ kvbuf, float* __restrict__ ksum) {
    const int bh  = blockIdx.y;
    const int tid = threadIdx.x;
    const int di = tid >> 4, ej = tid & 15;
    const int d0 = di * 4, e0 = ej * 4;
    float acc[4][4] = {};
    float sk[4] = {};
    const size_t base = ((size_t)bh * TOKENS + blockIdx.x * 1024) * HDIM;
#pragma unroll 4
    for (int n = 0; n < 1024; ++n) {
        union { unsigned long long u; __bf16 h[4]; } kk, vv;
        kk.u = *(const unsigned long long*)(kh + base + (size_t)n * HDIM + d0);
        vv.u = *(const unsigned long long*)(vh + base + (size_t)n * HDIM + e0);
        float kf[4], vf[4];
#pragma unroll
        for (int a = 0; a < 4; ++a) { kf[a] = (float)kk.h[a]; vf[a] = (float)vv.h[a]; }
#pragma unroll
        for (int a = 0; a < 4; ++a)
#pragma unroll
            for (int c = 0; c < 4; ++c) acc[a][c] += kf[a] * vf[c];
        if (ej == 0)
#pragma unroll
            for (int a = 0; a < 4; ++a) sk[a] += kf[a];
    }
#pragma unroll
    for (int a = 0; a < 4; ++a)
#pragma unroll
        for (int c = 0; c < 4; ++c)
            atomicAdd(&kvbuf[(size_t)bh * 4096 + (d0 + a) * HDIM + e0 + c], acc[a][c]);
    if (ej == 0)
#pragma unroll
        for (int a = 0; a < 4; ++a) atomicAdd(&ksum[bh * HDIM + d0 + a], sk[a]);
}

// ======================= out = (q @ kv) * z, WMMA ==========================
// grid.x = TOKENS/128, grid.y = b*h.  attn: [b*TOKENS + n][h*64 + e] bf16.
__global__ __launch_bounds__(256)
void attn_out_kernel(const __bf16* __restrict__ qh, const float* __restrict__ kvbuf,
                     const float* __restrict__ ksum, __bf16* __restrict__ attn) {
    __shared__ __bf16 qs[128][64];   // stride 128B: b128 rows stay 16B aligned
    __shared__ __bf16 kvt[64][66];   // kv transposed: kvt[e][d]
    __shared__ float  zb[128];

    const int tid  = threadIdx.x;
    const int lane = tid & 31, wave = tid >> 5;
    const int lrow = lane & 15, lhi = lane >> 4;
    const int bh = blockIdx.y, b = bh >> 3, h = bh & 7;
    const int n0 = blockIdx.x * 128;
    const size_t qbase = ((size_t)bh * TOKENS + n0) * HDIM;

    // stage q tile (128x64 bf16 = 16KB)
#pragma unroll
    for (int it = 0; it < 4; ++it) {
        const int idx = (tid + it * 256) * 8;
        const int r = idx >> 6, c = idx & 63;
        const __bf16* src = qh + qbase + (size_t)r * HDIM + c;
#if USE_ASYNC_LDS
        __builtin_amdgcn_global_load_async_to_lds_b128(
            AS1G(src), AS3L(&qs[r][c]), 0, 0);
#else
        const unsigned long long* s2 = (const unsigned long long*)src;
        unsigned long long d0 = s2[0], d1 = s2[1];
        *(unsigned int*)&qs[r][c]     = (unsigned int)d0;
        *(unsigned int*)&qs[r][c + 2] = (unsigned int)(d0 >> 32);
        *(unsigned int*)&qs[r][c + 4] = (unsigned int)d1;
        *(unsigned int*)&qs[r][c + 6] = (unsigned int)(d1 >> 32);
#endif
    }
    // stage kv (f32 -> bf16, transposed)
#pragma unroll
    for (int it = 0; it < 16; ++it) {
        const int idx = tid + it * 256;
        const int d = idx >> 6, e = idx & 63;
        kvt[e][d] = (__bf16)kvbuf[(size_t)bh * 4096 + idx];
    }
#if USE_ASYNC_LDS
    __builtin_amdgcn_s_wait_asynccnt(0);
#endif
    __syncthreads();

    if (tid < 128) {
        float s = 0.f;
#pragma unroll
        for (int c = 0; c < HDIM; ++c) s += (float)qs[tid][c] * ksum[bh * HDIM + c];
        zb[tid] = 1.f / (s + 1e-6f);
    }
    __syncthreads();

    const int r0 = wave * 16;   // 8 waves x 16 rows = 128 rows
    v8f acc[4];
#pragma unroll
    for (int tn = 0; tn < 4; ++tn)
#pragma unroll
        for (int i = 0; i < 8; ++i) acc[tn][i] = 0.f;

    union Uv { unsigned int u[8]; v16bf v; };
#pragma unroll
    for (int ks = 0; ks < 2; ++ks) {
        Uv afr;
        const int ar = r0 + lrow;
        const int kb = ks * 32 + lhi * 16;
#pragma unroll
        for (int v = 0; v < 8; ++v) {
            const int kk = kb + v * 2;
            afr.u[v] = *(const unsigned int*)&qs[ar][kk];
        }
#pragma unroll
        for (int tn = 0; tn < 4; ++tn) {
            Uv bfr;
            const int bc = tn * 16 + lrow;
            const int kbB = ks * 32 + lhi * 16;
#pragma unroll
            for (int v = 0; v < 8; ++v)
                bfr.u[v] = *(const unsigned int*)&kvt[bc][kbB + v * 2];
            acc[tn] = __builtin_amdgcn_wmma_f32_16x16x32_bf16(
                false, afr.v, false, bfr.v, (short)0, acc[tn], false, false);
        }
    }

#pragma unroll
    for (int tn = 0; tn < 4; ++tn) {
        const int e = tn * 16 + lrow;
#pragma unroll
        for (int i = 0; i < 8; ++i) {
            const int m = r0 + lhi * 8 + i;
            const float val = acc[tn][i] * zb[m];
            attn[((size_t)(b * TOKENS + n0 + m)) * CH + h * HDIM + e] = (__bf16)val;
        }
    }
}

// ======================= host orchestration ================================
extern "C" void kernel_launch(void* const* d_in, const int* in_sizes, int n_in,
                              void* d_out, int out_size, void* d_ws, size_t ws_size,
                              hipStream_t stream) {
    const float* x      = (const float*)d_in[0];
    const float* w_qkv  = (const float*)d_in[1];
    const float* b_qkv  = (const float*)d_in[2];
    const float* w_proj = (const float*)d_in[3];
    const float* b_proj = (const float*)d_in[4];
    float* out = (float*)d_out;

    char* ws = (char*)d_ws;
    size_t off = 0;
    auto alloc = [&](size_t bytes) -> void* {
        void* p = ws + off;
        off = (off + bytes + 255) & ~(size_t)255;
        return p;
    };

    __bf16* xh    = (__bf16*)alloc((size_t)BN_TOK * CH * 2);
    __bf16* wqh   = (__bf16*)alloc((size_t)CH * 3 * CH * 2);
    __bf16* wph   = (__bf16*)alloc((size_t)CH * CH * 2);
    __bf16* qkvh  = (__bf16*)alloc((size_t)BN_TOK * 3 * CH * 2);
    __bf16* qh    = (__bf16*)alloc((size_t)BN_TOK * CH * 2);
    __bf16* kh    = (__bf16*)alloc((size_t)BN_TOK * CH * 2);
    __bf16* vh    = (__bf16*)alloc((size_t)BN_TOK * CH * 2);
    __bf16* attn  = (__bf16*)alloc((size_t)BN_TOK * CH * 2);
    float*  kvbuf = (float*)alloc((size_t)BATCH * HEADS * HDIM * HDIM * 4);
    float*  ksum  = (float*)alloc((size_t)BATCH * HEADS * HDIM * 4);

    // 1) convert operands to bf16
    {
        int n = BN_TOK * CH;
        cvt_f32_bf16<<<(n + 255) / 256, 256, 0, stream>>>(x, xh, n);
        n = CH * 3 * CH;
        cvt_f32_bf16<<<(n + 255) / 256, 256, 0, stream>>>(w_qkv, wqh, n);
        n = CH * CH;
        cvt_f32_bf16<<<(n + 255) / 256, 256, 0, stream>>>(w_proj, wph, n);
    }

    // 2) QKV GEMM: [65536,512] @ [512,1536] + bias -> bf16
    gemm_bf16_kernel<false><<<dim3((3 * CH) / 128, BN_TOK / 64), 256, 0, stream>>>(
        xh, wqh, b_qkv, qkvh, BN_TOK, 3 * CH, CH);

    // 3) RoPE + elu+1, split into head-major q/k/v
    rope_elu_kernel<<<(BN_TOK * HEADS * 32) / 256, 256, 0, stream>>>(qkvh, qh, kh, vh);

    // 4) kv outer-product and k-sum reductions (zero accumulators first)
    (void)hipMemsetAsync(kvbuf, 0, (size_t)BATCH * HEADS * HDIM * HDIM * 4, stream);
    (void)hipMemsetAsync(ksum, 0, (size_t)BATCH * HEADS * HDIM * 4, stream);
    kv_reduce_kernel<<<dim3(TOKENS / 1024, BATCH * HEADS), 256, 0, stream>>>(
        kh, vh, kvbuf, ksum);

    // 5) out = (q @ kv) * z  -> attn [bn][512] bf16
    attn_out_kernel<<<dim3(TOKENS / 128, BATCH * HEADS), 256, 0, stream>>>(
        qh, kvbuf, ksum, attn);

    // 6) projection GEMM: [65536,512] @ [512,512] + bias -> f32 d_out
    gemm_bf16_kernel<true><<<dim3(CH / 128, BN_TOK / 64), 256, 0, stream>>>(
        attn, wph, b_proj, out, BN_TOK, CH, CH);
}
